// EncoderBlock_58342835749696
// MI455X (gfx1250) — matrix operand order repaired and checked by reading, not compiled, not run on hardware
//
#include <hip/hip_runtime.h>
#include <hip/hip_bf16.h>

// ---------------- problem constants ----------------
#define SEQ   8192
#define HID   128
#define NCONV 4
#define KS    7
#define LNEPS 1e-5f
#define QSCALE 0.08838834764831845f   // 1/sqrt(128)
#define KSPLIT 2                      // attention key-dimension split
#define KCHUNK (SEQ / KSPLIT)

typedef __attribute__((ext_vector_type(16))) _Float16 v16h;
typedef __attribute__((ext_vector_type(8)))  float    v8f;
typedef __attribute__((ext_vector_type(4)))  _Float16 v4h;

// ---------------- WMMA fragment helpers (CDNA5 gfx1250, wave32) ----------------
// A (16x32 f16): lane l: M = l&15, hs = l>>4. VGPR j holds K pair:
//   K = (j>>2)*16 + hs*8 + (j&3)*2   (ISA 7.12.2 "16-bit A-Matrix 16x32")
__device__ __forceinline__ int a_kidx(int j, int hs) {
  return ((j >> 2) << 4) | (hs << 3) | ((j & 3) << 1);
}
// B (32x16 f16): lane l: N = l&15, hs = l>>4. VGPR j holds K pair:
//   K = (j>>2)*16 + hs*8 + (j&3)*2   (same K order as A)
__device__ __forceinline__ int b_kidx(int j, int hs) {
  return ((j >> 2) << 4) | (hs << 3) | ((j & 3) << 1);
}

__device__ __forceinline__ v8f wmma_f16(v16h a, v16h b, v8f c) {
  return __builtin_amdgcn_wmma_f32_16x16x32_f16(
      /*neg_a=*/false, a, /*neg_b=*/false, b,
      /*c_mod=*/(short)0, c, /*reuse_a=*/false, /*reuse_b=*/false);
}

// A fragment from row-major f16 [rows x ld], rows (row0..row0+15), cols (col0..col0+31)
__device__ __forceinline__ v16h load_a(const _Float16* __restrict__ p,
                                       long row0, int col0, int ld) {
  int lane = threadIdx.x & 31;
  int m = lane & 15, hs = lane >> 4;
  const _Float16* rp = p + (row0 + m) * (long)ld + col0;
  v16h a;
#pragma unroll
  for (int j = 0; j < 8; ++j) {
    int kb = a_kidx(j, hs);
    a[2 * j]     = rp[kb];
    a[2 * j + 1] = rp[kb + 1];
  }
  return a;
}

// B fragment where B(k,n) = p[(n0+n)*ld + k0+k]  (source N-major, K contiguous)
__device__ __forceinline__ v16h load_b_kt(const _Float16* __restrict__ p,
                                          int k0, int n0, int ld) {
  int lane = threadIdx.x & 31;
  int n = lane & 15, hs = lane >> 4;
  const _Float16* rp = p + (long)(n0 + n) * ld + k0;
  v16h b;
#pragma unroll
  for (int j = 0; j < 8; ++j) {
    int kb = b_kidx(j, hs);
    b[2 * j]     = rp[kb];
    b[2 * j + 1] = rp[kb + 1];
  }
  return b;
}

// ---------------- kernel 0: weight prep (f32 -> f16, conv transposed to [i][k][o][c]) ----
__global__ void prep_kernel(const float* __restrict__ convw,
                            const float* __restrict__ Wq, const float* __restrict__ Wk,
                            const float* __restrict__ Wv, const float* __restrict__ Wff,
                            _Float16* __restrict__ wt16,
                            _Float16* __restrict__ wq16, _Float16* __restrict__ wk16,
                            _Float16* __restrict__ wv16, _Float16* __restrict__ wff16) {
  int idx = blockIdx.x * 256 + threadIdx.x;
  const int NC = NCONV * KS * HID * HID;  // 458752
  if (idx < NC) {
    int c = idx & 127;
    int t = idx >> 7;
    int o = t & 127;
    t >>= 7;
    int k = t % KS;
    int i = t / KS;
    // src: conv_w[i][o][c][k]  (shape [NCONV,H,H,K])
    wt16[idx] = (_Float16)convw[((long)(i * HID + o) * HID + c) * KS + k];
  } else {
    int j = idx - NC;
    if (j < 16384)      wq16[j]          = (_Float16)Wq[j];
    else if (j < 32768) wk16[j - 16384]  = (_Float16)Wk[j - 16384];
    else if (j < 49152) wv16[j - 32768]  = (_Float16)Wv[j - 32768];
    else                wff16[j - 49152] = (_Float16)Wff[j - 49152];
  }
}

// ---------------- kernel 1: h = x + positional encoding ----------------
__global__ void posadd_kernel(const float* __restrict__ x, float* __restrict__ h) {
  int idx = blockIdx.x * 256 + threadIdx.x;   // over SEQ*HID
  int l = idx >> 7;
  int i = idx & 127;
  float fi2 = (float)((i >> 1) << 1);
  // pos / 10000^(fi2/H)  ==  pos * exp(-(fi2/H) * ln(10000))
  float angle = (float)l * __expf(-(fi2 * (1.0f / (float)HID)) * 9.210340371976184f);
  float pe = (i & 1) ? __cosf(angle) : __sinf(angle);
  h[idx] = x[idx] + pe;
}

// ---------------- kernel 1b: zero the 16-row pads around the LN buffer ----------------
__global__ void pad_kernel(_Float16* __restrict__ lnp) {
  int idx = blockIdx.x * 256 + threadIdx.x;   // 2*16*HID = 4096 elements
  _Float16 z = (_Float16)0.f;
  if (idx < 16 * HID) lnp[idx] = z;
  else                lnp[(long)(SEQ + 16) * HID + (idx - 16 * HID)] = z;
}

// ---------------- kernel 2: LayerNorm rows -> f16 (writes interior of padded buf) ----
// one wave per row (H=128 -> 4 f32 per lane), 8 rows per 256-thread block
__global__ void ln_kernel(const float* __restrict__ h, _Float16* __restrict__ out) {
  int wave = threadIdx.x >> 5;
  int lane = threadIdx.x & 31;
  long row = (long)blockIdx.x * 8 + wave;
  const float* rp = h + row * HID;
  float4 v = *(const float4*)(rp + lane * 4);
  float s = v.x + v.y + v.z + v.w;
#pragma unroll
  for (int d = 1; d <= 16; d <<= 1) s += __shfl_xor(s, d, 32);
  float mu = s * (1.0f / (float)HID);
  float d0 = v.x - mu, d1 = v.y - mu, d2 = v.z - mu, d3 = v.w - mu;
  float q = d0 * d0 + d1 * d1 + d2 * d2 + d3 * d3;
#pragma unroll
  for (int d = 1; d <= 16; d <<= 1) q += __shfl_xor(q, d, 32);
  float rs = rsqrtf(q * (1.0f / (float)HID) + LNEPS);
  v4h o;
  o[0] = (_Float16)(d0 * rs);
  o[1] = (_Float16)(d1 * rs);
  o[2] = (_Float16)(d2 * rs);
  o[3] = (_Float16)(d3 * rs);
  *(v4h*)(out + row * HID + lane * 4) = o;
}

// ---------------- kernel 3: conv1d K=7 SAME as 7 fused K-GEMMs + residual ----------------
// one wave -> 32 rows x 32 out-channels (2x2 WMMA tiles, 2 b128 loads per WMMA);
// block = 4 waves (128 rows x 32 channels).
// ln16 points at the interior of a zero-padded buffer, so rows [-3, SEQ+2] are valid.
__global__ void conv_kernel(const _Float16* __restrict__ ln16,
                            const _Float16* __restrict__ wt16,  // [i][k][o][c] f16
                            const float* __restrict__ cb,       // [NCONV][H]
                            float* __restrict__ h, int ci) {
  int wave = threadIdx.x >> 5, lane = threadIdx.x & 31;
  int l0 = blockIdx.x * 128 + wave * 32;
  int o0 = blockIdx.y * 32;
  const _Float16* wbase = wt16 + (long)ci * KS * HID * HID;
  v8f a00 = {}, a01 = {}, a10 = {}, a11 = {};
  for (int k = 0; k < KS; ++k) {
    const _Float16* wk = wbase + (long)k * HID * HID;
#pragma unroll
    for (int c0 = 0; c0 < HID; c0 += 32) {
      v16h fa0 = load_a(ln16, l0 + k - 3,      c0, HID);
      v16h fa1 = load_a(ln16, l0 + 16 + k - 3, c0, HID);
      v16h fb0 = load_b_kt(wk, c0, o0,      HID);
      v16h fb1 = load_b_kt(wk, c0, o0 + 16, HID);
      a00 = wmma_f16(fa0, fb0, a00);
      a01 = wmma_f16(fa0, fb1, a01);
      a10 = wmma_f16(fa1, fb0, a10);
      a11 = wmma_f16(fa1, fb1, a11);
    }
  }
  int n = lane & 15, hs = lane >> 4;
  float bias0 = cb[ci * HID + o0 + n];
  float bias1 = cb[ci * HID + o0 + 16 + n];
#pragma unroll
  for (int r = 0; r < 8; ++r) {
    long r0 = (long)(l0 + r + 8 * hs) * HID;
    long r1 = (long)(l0 + 16 + r + 8 * hs) * HID;
    h[r0 + o0 + n]      += a00[r] + bias0;
    h[r0 + o0 + 16 + n] += a01[r] + bias1;
    h[r1 + o0 + n]      += a10[r] + bias0;
    h[r1 + o0 + 16 + n] += a11[r] + bias1;
  }
}

// ---------------- kernel 4: QKV projections (grid.z picks q/k/v) ----------------
// 2x2 tiles per wave; q is pre-scaled by 1/sqrt(H); v is stored TRANSPOSED: vT[ch][seq]
__global__ void qkv_kernel(const _Float16* __restrict__ ln16,
                           const _Float16* __restrict__ wq, const _Float16* __restrict__ wk,
                           const _Float16* __restrict__ wv,
                           const float* __restrict__ bq, const float* __restrict__ bk,
                           const float* __restrict__ bv,
                           _Float16* __restrict__ qo, _Float16* __restrict__ ko,
                           _Float16* __restrict__ vTo) {
  int which = blockIdx.z;
  const _Float16* W  = (which == 0) ? wq : (which == 1) ? wk : wv;
  const float*    bb = (which == 0) ? bq : (which == 1) ? bk : bv;
  float scale = (which == 0) ? QSCALE : 1.0f;
  int wave = threadIdx.x >> 5, lane = threadIdx.x & 31;
  long l0 = (long)blockIdx.x * 128 + wave * 32;
  int o0 = blockIdx.y * 32;
  v8f a00 = {}, a01 = {}, a10 = {}, a11 = {};
#pragma unroll
  for (int c0 = 0; c0 < HID; c0 += 32) {
    v16h fa0 = load_a(ln16, l0,      c0, HID);
    v16h fa1 = load_a(ln16, l0 + 16, c0, HID);
    v16h fb0 = load_b_kt(W, c0, o0,      HID);   // B(c,o) = W[o*H + c]  (== W.T)
    v16h fb1 = load_b_kt(W, c0, o0 + 16, HID);
    a00 = wmma_f16(fa0, fb0, a00);
    a01 = wmma_f16(fa0, fb1, a01);
    a10 = wmma_f16(fa1, fb0, a10);
    a11 = wmma_f16(fa1, fb1, a11);
  }
  int n = lane & 15, hs = lane >> 4;
  float bias0 = bb[o0 + n];
  float bias1 = bb[o0 + 16 + n];
  if (which == 2) {
#pragma unroll
    for (int r = 0; r < 8; ++r) {
      long s0 = l0 + r + 8 * hs, s1 = l0 + 16 + r + 8 * hs;
      vTo[(long)(o0 + n) * SEQ + s0]      = (_Float16)(a00[r] + bias0);
      vTo[(long)(o0 + 16 + n) * SEQ + s0] = (_Float16)(a01[r] + bias1);
      vTo[(long)(o0 + n) * SEQ + s1]      = (_Float16)(a10[r] + bias0);
      vTo[(long)(o0 + 16 + n) * SEQ + s1] = (_Float16)(a11[r] + bias1);
    }
  } else {
    _Float16* out = (which == 0) ? qo : ko;
#pragma unroll
    for (int r = 0; r < 8; ++r) {
      long r0 = (l0 + r + 8 * hs) * HID, r1 = (l0 + 16 + r + 8 * hs) * HID;
      out[r0 + o0 + n]      = (_Float16)((a00[r] + bias0) * scale);
      out[r0 + o0 + 16 + n] = (_Float16)((a01[r] + bias1) * scale);
      out[r1 + o0 + n]      = (_Float16)((a10[r] + bias0) * scale);
      out[r1 + o0 + 16 + n] = (_Float16)((a11[r] + bias1) * scale);
    }
  }
}

// ---------------- kernel 5: streaming (flash) attention, key-split partials ----------
// grid = (SEQ/32, KSPLIT), block = 2 waves. Each wave: 16 query rows, keys in tiles
// of 32, online softmax in f32. Writes unnormalized O plus per-row (m, l).
#define ATT_WAVES 2
__global__ void attn_kernel(const _Float16* __restrict__ q16,
                            const _Float16* __restrict__ k16,
                            const _Float16* __restrict__ vT16,
                            float* __restrict__ Opart,
                            float* __restrict__ Mpart,
                            float* __restrict__ Lpart) {
  __shared__ _Float16 lds[ATT_WAVES][16][40];   // per-wave P staging (C-layout -> A-layout)
  int wave = threadIdx.x >> 5, lane = threadIdx.x & 31;
  int ks = blockIdx.y;
  long l0 = (long)blockIdx.x * (16 * ATT_WAVES) + wave * 16;
  int m = lane & 15, hs = lane >> 4;
  _Float16 (*P)[40] = lds[wave];

  // resident Q fragments (q already scaled by 1/sqrt(H))
  v16h qa[4];
#pragma unroll
  for (int c = 0; c < 4; ++c) qa[c] = load_a(q16, l0, c * 32, HID);

  v8f O[8];
  float mrow[8], lrow[8];
  v8f vzero = {};
#pragma unroll
  for (int t = 0; t < 8; ++t) { O[t] = vzero; mrow[t] = -1e30f; lrow[t] = 0.0f; }

  int jend = (ks + 1) * KCHUNK;
  for (int j0 = ks * KCHUNK; j0 < jend; j0 += 32) {
    // S = Q @ K^T for 32 keys: two 16x16 tiles
    v8f S0 = vzero, S1 = vzero;
#pragma unroll
    for (int c = 0; c < 4; ++c) {
      S0 = wmma_f16(qa[c], load_b_kt(k16, c * 32, j0,      HID), S0);
      S1 = wmma_f16(qa[c], load_b_kt(k16, c * 32, j0 + 16, HID), S1);
    }
    // online softmax per row r (C layout: lane=N, VGPR r -> M=r+8*hs)
#pragma unroll
    for (int r = 0; r < 8; ++r) {
      float mx = fmaxf(S0[r], S1[r]);
#pragma unroll
      for (int d = 1; d <= 8; d <<= 1) mx = fmaxf(mx, __shfl_xor(mx, d, 32));
      float mnew  = fmaxf(mrow[r], mx);
      float alpha = __expf(mrow[r] - mnew);
      float p0 = __expf(S0[r] - mnew);
      float p1 = __expf(S1[r] - mnew);
      float rs = p0 + p1;
#pragma unroll
      for (int d = 1; d <= 8; d <<= 1) rs += __shfl_xor(rs, d, 32);
      lrow[r] = lrow[r] * alpha + rs;
      mrow[r] = mnew;
#pragma unroll
      for (int t = 0; t < 8; ++t) O[t][r] *= alpha;
      int mr = r + 8 * hs;
      P[mr][m]      = (_Float16)p0;
      P[mr][16 + m] = (_Float16)p1;
    }
    // re-read P as an A fragment (wave-local LDS: DS ops are in-order per wave)
    v16h pa;
#pragma unroll
    for (int j = 0; j < 8; ++j) {
      int kb = a_kidx(j, hs);
      pa[2 * j]     = P[m][kb];
      pa[2 * j + 1] = P[m][kb + 1];
    }
    // O += P @ V   (vT is channel-major, so B fragments are K-contiguous b128 loads)
#pragma unroll
    for (int t = 0; t < 8; ++t)
      O[t] = wmma_f16(pa, load_b_kt(vT16, j0, t * 16, SEQ), O[t]);
  }

  // store unnormalized partials
#pragma unroll
  for (int r = 0; r < 8; ++r) {
    long rowg = l0 + r + 8 * hs;
    float* Op = Opart + ((long)ks * SEQ + rowg) * HID;
#pragma unroll
    for (int t = 0; t < 8; ++t) Op[t * 16 + m] = O[t][r];
    if (m == 0) {
      Mpart[(long)ks * SEQ + rowg] = mrow[r];
      Lpart[(long)ks * SEQ + rowg] = lrow[r];
    }
  }
}

// ---------------- kernel 5b: merge key-split partials + residual ----------------
__global__ void attn_merge_kernel(const float* __restrict__ Opart,
                                  const float* __restrict__ Mpart,
                                  const float* __restrict__ Lpart,
                                  float* __restrict__ h) {
  int wave = threadIdx.x >> 5, lane = threadIdx.x & 31;
  long row = (long)blockIdx.x * 8 + wave;
  float m0 = Mpart[row],       m1 = Mpart[SEQ + row];
  float l0 = Lpart[row],       l1 = Lpart[SEQ + row];
  float M  = fmaxf(m0, m1);
  float w0 = __expf(m0 - M), w1 = __expf(m1 - M);
  float inv = 1.0f / (l0 * w0 + l1 * w1);
  const float* O0 = Opart + row * HID + lane * 4;
  const float* O1 = Opart + (long)SEQ * HID + row * HID + lane * 4;
  float4 a = *(const float4*)O0;
  float4 b = *(const float4*)O1;
  long base = row * HID + lane * 4;
  h[base + 0] += (a.x * w0 + b.x * w1) * inv;
  h[base + 1] += (a.y * w0 + b.y * w1) * inv;
  h[base + 2] += (a.z * w0 + b.z * w1) * inv;
  h[base + 3] += (a.w * w0 + b.w * w1) * inv;
}

// ---------------- kernel 6: FFN (relu(LN(h)@Wff.T + b) + h) -> d_out ----------------
// 2x2 tiles per wave, same blocking as qkv
__global__ void ffn_kernel(const _Float16* __restrict__ ln16,
                           const _Float16* __restrict__ wff,
                           const float* __restrict__ bff,
                           const float* __restrict__ hres,
                           float* __restrict__ out) {
  int wave = threadIdx.x >> 5, lane = threadIdx.x & 31;
  long l0 = (long)blockIdx.x * 128 + wave * 32;
  int o0 = blockIdx.y * 32;
  v8f a00 = {}, a01 = {}, a10 = {}, a11 = {};
#pragma unroll
  for (int c0 = 0; c0 < HID; c0 += 32) {
    v16h fa0 = load_a(ln16, l0,      c0, HID);
    v16h fa1 = load_a(ln16, l0 + 16, c0, HID);
    v16h fb0 = load_b_kt(wff, c0, o0,      HID);
    v16h fb1 = load_b_kt(wff, c0, o0 + 16, HID);
    a00 = wmma_f16(fa0, fb0, a00);
    a01 = wmma_f16(fa0, fb1, a01);
    a10 = wmma_f16(fa1, fb0, a10);
    a11 = wmma_f16(fa1, fb1, a11);
  }
  int n = lane & 15, hs = lane >> 4;
  float bias0 = bff[o0 + n];
  float bias1 = bff[o0 + 16 + n];
#pragma unroll
  for (int r = 0; r < 8; ++r) {
    long r0 = (l0 + r + 8 * hs) * HID, r1 = (l0 + 16 + r + 8 * hs) * HID;
    out[r0 + o0 + n]      = fmaxf(a00[r] + bias0, 0.0f) + hres[r0 + o0 + n];
    out[r0 + o0 + 16 + n] = fmaxf(a01[r] + bias1, 0.0f) + hres[r0 + o0 + 16 + n];
    out[r1 + o0 + n]      = fmaxf(a10[r] + bias0, 0.0f) + hres[r1 + o0 + n];
    out[r1 + o0 + 16 + n] = fmaxf(a11[r] + bias1, 0.0f) + hres[r1 + o0 + 16 + n];
  }
}

// ---------------- launcher ----------------
#define WSALLOC(ptrtype, name, bytes) \
  ptrtype name = (ptrtype)(ws + ws_off); ws_off += (((size_t)(bytes)) + 255) & ~(size_t)255;

extern "C" void kernel_launch(void* const* d_in, const int* in_sizes, int n_in,
                              void* d_out, int out_size, void* d_ws, size_t ws_size,
                              hipStream_t stream) {
  const float* x     = (const float*)d_in[0];
  const float* convw = (const float*)d_in[1];
  const float* convb = (const float*)d_in[2];
  const float* Wq  = (const float*)d_in[3];
  const float* bq  = (const float*)d_in[4];
  const float* Wk  = (const float*)d_in[5];
  const float* bk  = (const float*)d_in[6];
  const float* Wv  = (const float*)d_in[7];
  const float* bv  = (const float*)d_in[8];
  const float* Wff = (const float*)d_in[9];
  const float* bff = (const float*)d_in[10];
  float* out = (float*)d_out;

  // workspace carve-out (~22 MB total)
  char* ws = (char*)d_ws;
  size_t ws_off = 0;
  WSALLOC(float*,     h,     SEQ * HID * 4)
  WSALLOC(_Float16*,  lnp,   (SEQ + 32) * HID * 2)     // 16-row zero pads on each side
  WSALLOC(_Float16*,  q16,   SEQ * HID * 2)
  WSALLOC(_Float16*,  k16,   SEQ * HID * 2)
  WSALLOC(_Float16*,  vT16,  SEQ * HID * 2)            // [HID][SEQ]
  WSALLOC(_Float16*,  wt16,  NCONV * KS * HID * HID * 2)
  WSALLOC(_Float16*,  wq16,  HID * HID * 2)
  WSALLOC(_Float16*,  wk16,  HID * HID * 2)
  WSALLOC(_Float16*,  wv16,  HID * HID * 2)
  WSALLOC(_Float16*,  wff16, HID * HID * 2)
  WSALLOC(float*,     Opart, (size_t)KSPLIT * SEQ * HID * 4)
  WSALLOC(float*,     Mpart, KSPLIT * SEQ * 4)
  WSALLOC(float*,     Lpart, KSPLIT * SEQ * 4)
  _Float16* ln16 = lnp + 16 * HID;                     // interior (row 0)

  // 0) weight prep + LN pad zeroing
  prep_kernel<<<2048, 256, 0, stream>>>(convw, Wq, Wk, Wv, Wff,
                                        wt16, wq16, wk16, wv16, wff16);
  pad_kernel<<<16, 256, 0, stream>>>(lnp);

  // 1) h = x + PE
  posadd_kernel<<<(SEQ * HID) / 256, 256, 0, stream>>>(x, h);

  // 2) conv stack
  for (int i = 0; i < NCONV; ++i) {
    ln_kernel<<<SEQ / 8, 256, 0, stream>>>(h, ln16);
    conv_kernel<<<dim3(SEQ / 128, HID / 32), 128, 0, stream>>>(ln16, wt16, convb, h, i);
  }

  // 3) attention
  ln_kernel<<<SEQ / 8, 256, 0, stream>>>(h, ln16);
  qkv_kernel<<<dim3(SEQ / 128, HID / 32, 3), 128, 0, stream>>>(
      ln16, wq16, wk16, wv16, bq, bk, bv, q16, k16, vT16);
  attn_kernel<<<dim3(SEQ / (16 * ATT_WAVES), KSPLIT), 32 * ATT_WAVES, 0, stream>>>(
      q16, k16, vT16, Opart, Mpart, Lpart);
  attn_merge_kernel<<<SEQ / 8, 256, 0, stream>>>(Opart, Mpart, Lpart, h);

  // 4) FFN -> d_out
  ln_kernel<<<SEQ / 8, 256, 0, stream>>>(h, ln16);
  ffn_kernel<<<dim3(SEQ / 128, HID / 32), 128, 0, stream>>>(ln16, wff16, bff, h, out);
}
